// AudioMamba_32865089749678
// MI455X (gfx1250) — matrix operand order repaired and checked by reading, not compiled, not run on hardware
//
#include <hip/hip_runtime.h>
#include <hip/hip_bf16.h>
#include <cstdint>

// ---------------- model dims ----------------
#define D_     512
#define DI_    1024
#define DS_    16
#define DR_    32
#define KK_    4
#define DEPTH_ 2
#define B_     8
#define L_     513
#define LT_    77
#define H_     8
#define DH_    64
#define M_     (B_ * L_)    // 4104 sequence rows
#define M2_    (B_ * LT_)   // 616 text rows

// ---------------- CDNA5 WMMA / TDM types ----------------
typedef __attribute__((ext_vector_type(16))) __bf16 v16bf;
typedef __attribute__((ext_vector_type(8)))  float  v8f;
typedef __attribute__((ext_vector_type(4)))  unsigned int u32x4;
typedef __attribute__((ext_vector_type(8)))  int i32x8;
typedef __attribute__((ext_vector_type(4)))  int i32x4;

union FragBF { uint4 q[2]; v16bf v; };
union AccU   { v8f v; float f[8]; };

__device__ __forceinline__ unsigned short f2bf(float f) {
  unsigned int u = __float_as_uint(f);
  unsigned int r = u + 0x7FFFu + ((u >> 16) & 1u);   // round-to-nearest-even
  return (unsigned short)(r >> 16);
}
__device__ __forceinline__ float siluf(float x) { return x / (1.0f + __expf(-x)); }

// ---------------- generic f32 -> bf16 cast ----------------
__global__ void cvt_bf16_kernel(const float* __restrict__ src,
                                unsigned short* __restrict__ dst, int n) {
  int i = blockIdx.x * 256 + threadIdx.x;
  if (i < n) dst[i] = f2bf(src[i]);
}

// ---------------- WMMA bf16 GEMM: C = A(MxK) * B(KxN), fp32 accumulate ----
// 128x128 tile / workgroup; 8 wave32 waves as 4x2; each wave 32x64 = 2x4 frags.
// A tiles are staged by the Tensor Data Mover (async DMA -> LDS, TENSORcnt),
// double-buffered so the DMA of tile k+1 overlaps the WMMAs of tile k.
#define TM 128
#define TN 128
#define TK 32
#define LSTR 40   // LDS K-stride (32 + 8 pad) in bf16 elems -> 80B rows

// Issue one TDM 2D-tile load: tile (TM rows x TK bf16) of row-major A (M x K)
// starting at (m0, k0) into LDS at byte offset ldsOff, with hardware padding
// of 16B after every 64B row (reproducing the LSTR=40 layout) and hardware
// zero-fill for rows beyond (Mdim - m0).  D# layout per CDNA5 ISA ch.8.
__device__ __forceinline__ void tdm_load_tileA(const unsigned short* A, int Mdim,
                                               int Kdim, int m0, int k0,
                                               unsigned int ldsOff) {
  unsigned long long ga =
      (unsigned long long)(size_t)A + (((size_t)m0 * (size_t)Kdim + (size_t)k0) << 1);
  unsigned int mrem = (unsigned int)(Mdim - m0);  // rows present; rest zero-fill
  u32x4 g0;
  g0.x = 1u;                                     // count=1 valid user descriptor
  g0.y = ldsOff;                                 // lds_addr (bytes)
  g0.z = (unsigned int)ga;                       // global_addr[31:0]
  g0.w = (unsigned int)((ga >> 32) & 0x01FFFFFFu) | (2u << 30);  // [56:32] | type=2
  i32x8 g1;
  g1[0] = (int)((1u << 16)      // data_size = 1 -> 2 bytes
              | (1u << 20)      // pad_enable
              | (3u << 22)      // pad_interval = 16 DWORDs (64B = one 32-elem row)
              | (3u << 25));    // pad_amount   = 4 DWORDs (16B -> 80B row stride)
  g1[1] = (int)(((unsigned)Kdim & 0xFFFFu) << 16);                 // tensor_dim0 lo
  g1[2] = (int)((((unsigned)Kdim >> 16) & 0xFFFFu) | ((mrem & 0xFFFFu) << 16));
  g1[3] = (int)(((mrem >> 16) & 0xFFFFu) | ((unsigned)TK << 16));  // tile_dim0 = 32
  g1[4] = (int)TM;                               // tile_dim1 = 128, tile_dim2 = 0
  g1[5] = (int)Kdim;                             // tensor_dim0_stride[31:0]
  g1[6] = 0;                                     // stride hi / dim1_stride lo
  g1[7] = 0;
  i32x4 gz;  gz[0] = 0; gz[1] = 0; gz[2] = 0; gz[3] = 0;      // groups 2/3 unused (2D)
  i32x8 gz8; for (int i = 0; i < 8; i++) gz8[i] = 0;          // trailing group, unused
  __builtin_amdgcn_tensor_load_to_lds(g0, g1, gz, gz, gz8, 0);
}

__global__ __launch_bounds__(256)
void gemm_bf16_wmma(const unsigned short* __restrict__ A,
                    const unsigned short* __restrict__ Bw,
                    float* __restrict__ C,
                    int Mdim, int Ndim, int Kdim) {
  __shared__ __align__(16) unsigned short lA[2][TM * LSTR];
  __shared__ __align__(16) unsigned short lB[TN * LSTR];

  const int tid   = threadIdx.x;
  const int wave  = tid >> 5;
  const int lane  = tid & 31;
  const int wm    = wave >> 1;        // 0..3
  const int wn    = wave & 1;         // 0..1
  const int lrow  = lane & 15;
  const int lhalf = lane >> 4;        // 0/1
  const int m0    = blockIdx.y * TM;
  const int n0    = blockIdx.x * TN;

  v8f acc[2][4];
  for (int i = 0; i < 2; i++)
    for (int j = 0; j < 4; j++)
      for (int r = 0; r < 8; r++) acc[i][j][r] = 0.0f;

  const int ktiles = Kdim / TK;   // K is always a multiple of 32 here

  // Kick off the TDM for the first A tile.
  if (wave == 0)
    tdm_load_tileA(A, Mdim, Kdim, m0, 0, (unsigned int)(size_t)&lA[0][0]);

  for (int kt = 0; kt < ktiles; kt++) {
    const int k0 = kt * TK;

    // ---- stage B tile transposed into [N][K] so frags are contiguous ----
    // Global side: two coalesced b128 loads (rows are 16-elem aligned since
    // every Ndim here is a multiple of 16); LDS side: 16-way b16 scatter.
    {
      int krow = tid >> 3, seg = tid & 7;        // krow 0..31, seg 0..7
      int gn0 = n0 + seg * 16;
      const unsigned short* src = Bw + (size_t)(k0 + krow) * Ndim + gn0;
      unsigned short tmp[16];
      if (gn0 + 16 <= Ndim) {
        *(uint4*)&tmp[0] = ((const uint4*)src)[0];
        *(uint4*)&tmp[8] = ((const uint4*)src)[1];
      } else {
#pragma unroll
        for (int j = 0; j < 16; j++)
          tmp[j] = (gn0 + j < Ndim) ? src[j] : (unsigned short)0;
      }
#pragma unroll
      for (int j = 0; j < 16; j++)
        lB[(seg * 16 + j) * LSTR + krow] = tmp[j];
    }
    // ---- software prefetch of next B tile (global_prefetch_b8) ----
    if (k0 + TK < Kdim)
      __builtin_prefetch(Bw + (size_t)(k0 + TK + (tid >> 3)) * Ndim + n0, 0, 1);

    // ---- pipeline the TDM: issue tile k+1, then wait for tile k ----
    if (wave == 0) {
      if (kt + 1 < ktiles) {
        tdm_load_tileA(A, Mdim, Kdim, m0, k0 + TK,
                       (unsigned int)(size_t)&lA[(kt + 1) & 1][0]);
        __builtin_amdgcn_s_wait_tensorcnt(1);  // tile k done (in-order), k+1 in flight
      } else {
        __builtin_amdgcn_s_wait_tensorcnt(0);
      }
    }
    __syncthreads();

    const unsigned short* lAc = &lA[kt & 1][0];

    // ---- fragment loads in contiguous k order (k = 16*lhalf + e), same for A and B ----
    // A lane: M = lane&15; K = e + (lane>=16 ? 16 : 0) -> contiguous 32B
    FragBF af[2];
#pragma unroll
    for (int i = 0; i < 2; i++) {
      const unsigned short* base = lAc + (wm * 32 + i * 16 + lrow) * LSTR;
      af[i].q[0] = *(const uint4*)(base + lhalf * 16);
      af[i].q[1] = *(const uint4*)(base + 8 + lhalf * 16);
    }
    // B lane: N = lane&15; K = e + (lane>=16 ? 16 : 0) -> contiguous 32B
    FragBF bf[4];
#pragma unroll
    for (int j = 0; j < 4; j++) {
      const unsigned short* base = &lB[(wn * 64 + j * 16 + lrow) * LSTR + lhalf * 16];
      bf[j].q[0] = ((const uint4*)base)[0];
      bf[j].q[1] = ((const uint4*)base)[1];
    }
#pragma unroll
    for (int i = 0; i < 2; i++)
#pragma unroll
      for (int j = 0; j < 4; j++)
        acc[i][j] = __builtin_amdgcn_wmma_f32_16x16x32_bf16(
            false, af[i].v, false, bf[j].v, (short)0, acc[i][j], false, false);
    __syncthreads();
  }

  // ---- store: C lane layout: (M = r + 8*lhalf, N = lane&15) per v8f elem r
#pragma unroll
  for (int i = 0; i < 2; i++)
#pragma unroll
    for (int j = 0; j < 4; j++) {
      AccU u; u.v = acc[i][j];
      int gn = n0 + wn * 64 + j * 16 + lrow;
      if (gn < Ndim) {
#pragma unroll
        for (int r = 0; r < 8; r++) {
          int gm = m0 + wm * 32 + i * 16 + lhalf * 8 + r;
          if (gm < Mdim) C[(size_t)gm * Ndim + gn] = u.f[r];
        }
      }
    }
}

// ---------------- fused residual-add + RMSNorm (+bf16 cast) ----------------
__global__ __launch_bounds__(256)
void prologue_kernel(const float* __restrict__ hs, float* __restrict__ R,
                     const float* __restrict__ normw, float* __restrict__ HN,
                     unsigned short* __restrict__ HNB, int first) {
  int row = blockIdx.x, tid = threadIdx.x;
  __shared__ float red[256];
  size_t base = (size_t)row * D_;
  float v0 = hs[base + tid], v1 = hs[base + 256 + tid];
  float r0, r1;
  if (first) { r0 = v0; r1 = v1; }
  else { r0 = R[base + tid] + v0; r1 = R[base + 256 + tid] + v1; }
  R[base + tid] = r0; R[base + 256 + tid] = r1;
  red[tid] = r0 * r0 + r1 * r1;
  __syncthreads();
  for (int s = 128; s > 0; s >>= 1) {
    if (tid < s) red[tid] += red[tid + s];
    __syncthreads();
  }
  float sc = rsqrtf(red[0] / (float)D_ + 1e-5f);
  float h0 = r0 * sc * normw[tid], h1 = r1 * sc * normw[256 + tid];
  HN[base + tid] = h0;            HN[base + 256 + tid] = h1;
  HNB[base + tid] = f2bf(h0);     HNB[base + 256 + tid] = f2bf(h1);
}

// ---------------- adaLN: mod = silu(c) @ adaln_w + adaln_b -----------------
__global__ __launch_bounds__(256)
void mod_kernel(const float* __restrict__ c, const float* __restrict__ aw,
                const float* __restrict__ ab, float* __restrict__ mod) {
  int b = blockIdx.x / 12, n = (blockIdx.x % 12) * 256 + threadIdx.x;
  __shared__ float cs[D_];
  for (int d = threadIdx.x; d < D_; d += 256) cs[d] = siluf(c[b * D_ + d]);
  __syncthreads();
  float acc = ab[n];
  for (int d = 0; d < D_; d++) acc += cs[d] * aw[(size_t)d * (6 * D_) + n];
  mod[b * (6 * D_) + n] = acc;
}

// ---------------- depthwise causal conv (K=4) + SiLU, dir-aware ------------
__global__ __launch_bounds__(256)
void conv_silu_kernel(const float* __restrict__ xz, const float* __restrict__ cw,
                      const float* __restrict__ cb, float* __restrict__ xc,
                      unsigned short* __restrict__ xcb, int dir) {
  int idx = blockIdx.x * 256 + threadIdx.x;
  if (idx >= M_ * DI_) return;
  int d = idx % DI_, t = (idx / DI_) % L_, b = idx / (DI_ * L_);
  float acc = cb[d];
#pragma unroll
  for (int k = 0; k < KK_; k++) {
    int tt = t - (KK_ - 1) + k;
    if (tt >= 0) {
      int st = dir ? (L_ - 1 - tt) : tt;  // bwd branch = conv over flipped xm
      acc += xz[((size_t)(b * L_ + st)) * (2 * DI_) + d] * cw[d * KK_ + k];
    }
  }
  float s = siluf(acc);
  xc[idx] = s;
  xcb[idx] = f2bf(s);
}

__global__ void prep_dtin_kernel(const float* __restrict__ proj,
                                 unsigned short* __restrict__ dtin) {
  int idx = blockIdx.x * 256 + threadIdx.x;
  if (idx >= M_ * DR_) return;
  dtin[idx] = f2bf(proj[(idx / DR_) * (DR_ + 2 * DS_) + (idx % DR_)]);
}

__global__ void softplus_kernel(float* __restrict__ dt, const float* __restrict__ bias) {
  int idx = blockIdx.x * 256 + threadIdx.x;
  if (idx >= M_ * DI_) return;
  float x = dt[idx] + bias[idx % DI_];
  dt[idx] = (x > 20.0f) ? x : log1pf(__expf(x));
}

// ---------------- selective scan: thread = (b, channel), LDS-shared B/C ----
__global__ __launch_bounds__(256)
void scan_kernel(const float* __restrict__ u, const float* __restrict__ dt,
                 const float* __restrict__ proj, const float* __restrict__ Alog,
                 const float* __restrict__ Dp, float* __restrict__ Y, int dir) {
  int b = blockIdx.x >> 2;
  int d = (blockIdx.x & 3) * 256 + threadIdx.x;
  float A[DS_], h[DS_];
#pragma unroll
  for (int s = 0; s < DS_; s++) { A[s] = -__expf(Alog[d * DS_ + s]); h[s] = 0.0f; }
  float Dd = Dp[d];
  __shared__ float BC[2 * DS_];
  for (int t = 0; t < L_; t++) {
    size_t row = (size_t)(b * L_ + t);
    if (threadIdx.x < 2 * DS_) BC[threadIdx.x] = proj[row * (DR_ + 2 * DS_) + DR_ + threadIdx.x];
    __syncthreads();
    float dtv = dt[row * DI_ + d];
    float uv  = u[row * DI_ + d];
    float du  = dtv * uv;
    float y   = uv * Dd;
#pragma unroll
    for (int s = 0; s < DS_; s++) {
      h[s] = __expf(dtv * A[s]) * h[s] + du * BC[s];
      y += h[s] * BC[DS_ + s];
    }
    int ot = dir ? (L_ - 1 - t) : t;            // bwd output is flipped back
    size_t o = (size_t)(b * L_ + ot) * DI_ + d;
    if (dir) Y[o] += y; else Y[o] = y;
    __syncthreads();
  }
}

// ---------------- y = (y_f + y_b) * silu(z) --------------------------------
__global__ void combine_kernel(const float* __restrict__ xz, float* __restrict__ Y,
                               unsigned short* __restrict__ yb) {
  int idx = blockIdx.x * 256 + threadIdx.x;
  if (idx >= M_ * DI_) return;
  int d = idx % DI_; size_t row = idx / DI_;
  float z = xz[row * (2 * DI_) + DI_ + d];
  float y = Y[idx] * siluf(z);
  Y[idx] = y; yb[idx] = f2bf(y);
}

// ---------------- x = hn + g_mba * 0.5 * mix -------------------------------
__global__ void xmod_kernel(const float* __restrict__ HN, const float* __restrict__ MIX,
                            const float* __restrict__ mod, float* __restrict__ X) {
  int idx = blockIdx.x * 256 + threadIdx.x;
  if (idx >= M_ * D_) return;
  int dd = idx % D_, b = idx / (D_ * L_);
  X[idx] = HN[idx] + mod[b * (6 * D_) + 2 * D_ + dd] * 0.5f * MIX[idx];
}

// ---------------- LN(no-affine) at token 256 + q projection ---------------
__global__ __launch_bounds__(256)
void lnq_kernel(const float* __restrict__ X, const float* __restrict__ mod,
                const float* __restrict__ wq, float* __restrict__ q) {
  int b = blockIdx.x, tid = threadIdx.x;
  __shared__ float xs[D_];
  __shared__ float red[256];
  const float* xr = X + ((size_t)(b * L_ + 256)) * D_;
  float v0 = xr[tid], v1 = xr[256 + tid];
  red[tid] = v0 + v1; __syncthreads();
  for (int s = 128; s > 0; s >>= 1) { if (tid < s) red[tid] += red[tid + s]; __syncthreads(); }
  float mean = red[0] / (float)D_; __syncthreads();
  float d0 = v0 - mean, d1 = v1 - mean;
  red[tid] = d0 * d0 + d1 * d1; __syncthreads();
  for (int s = 128; s > 0; s >>= 1) { if (tid < s) red[tid] += red[tid + s]; __syncthreads(); }
  float inv = rsqrtf(red[0] / (float)D_ + 1e-6f);
  const float* mb = mod + b * (6 * D_);
  xs[tid]       = d0 * inv * (1.0f + mb[4 * D_ + tid])       + mb[3 * D_ + tid];
  xs[256 + tid] = d1 * inv * (1.0f + mb[4 * D_ + 256 + tid]) + mb[3 * D_ + 256 + tid];
  __syncthreads();
  for (int n = tid; n < D_; n += 256) {
    float acc = 0.0f;
    for (int d = 0; d < D_; d++) acc += xs[d] * wq[(size_t)d * D_ + n];
    q[b * D_ + n] = acc;
  }
}

// ---------------- single-query attention per (b,h) -------------------------
__global__ __launch_bounds__(128)
void attn_kernel(const float* __restrict__ q, const float* __restrict__ K,
                 const float* __restrict__ V, float* __restrict__ O) {
  int b = blockIdx.x >> 3, h = blockIdx.x & 7, tid = threadIdx.x;
  __shared__ float qh[DH_];
  __shared__ float s1[128];
  __shared__ float p[128];
  if (tid < DH_) qh[tid] = q[b * D_ + h * DH_ + tid];
  __syncthreads();
  float sc = -1e30f;
  if (tid < LT_) {
    const float* kr = K + ((size_t)(b * LT_ + tid)) * D_ + h * DH_;
    float a = 0.0f;
    for (int j = 0; j < DH_; j++) a += qh[j] * kr[j];
    sc = a * 0.125f;  // 1/sqrt(64)
  }
  s1[tid] = sc; __syncthreads();
  for (int s = 64; s > 0; s >>= 1) { if (tid < s) s1[tid] = fmaxf(s1[tid], s1[tid + s]); __syncthreads(); }
  float mx = s1[0]; __syncthreads();
  float e = (tid < LT_) ? __expf(sc - mx) : 0.0f;
  s1[tid] = e; __syncthreads();
  for (int s = 64; s > 0; s >>= 1) { if (tid < s) s1[tid] += s1[tid + s]; __syncthreads(); }
  float denom = s1[0]; __syncthreads();
  p[tid] = e / denom; __syncthreads();
  if (tid < DH_) {
    float acc = 0.0f;
    for (int t = 0; t < LT_; t++)
      acc += p[t] * V[((size_t)(b * LT_ + t)) * D_ + h * DH_ + tid];
    O[b * D_ + h * DH_ + tid] = acc;
  }
}

// ---------------- o @ wo + wo_b --------------------------------------------
__global__ __launch_bounds__(256)
void oproj_kernel(const float* __restrict__ O, const float* __restrict__ wo,
                  const float* __restrict__ wob, float* __restrict__ O2) {
  int b = blockIdx.x >> 1, n = (blockIdx.x & 1) * 256 + threadIdx.x;
  __shared__ float os[D_];
  for (int d = threadIdx.x; d < D_; d += 256) os[d] = O[b * D_ + d];
  __syncthreads();
  float acc = wob[n];
  for (int d = 0; d < D_; d++) acc += os[d] * wo[(size_t)d * D_ + n];
  O2[b * D_ + n] = acc;
}

// ---------------- hs_next = x + g_msa * o ----------------------------------
__global__ void epilogue_kernel(const float* __restrict__ X, const float* __restrict__ mod,
                                const float* __restrict__ O2, float* __restrict__ OUT) {
  int idx = blockIdx.x * 256 + threadIdx.x;
  if (idx >= M_ * D_) return;
  int dd = idx % D_, b = idx / (D_ * L_);
  OUT[idx] = X[idx] + mod[b * (6 * D_) + 5 * D_ + dd] * O2[b * D_ + dd];
}

// ======================= host driver =======================
static inline int ceil_div(int a, int b) { return (a + b - 1) / b; }

extern "C" void kernel_launch(void* const* d_in, const int* in_sizes, int n_in,
                              void* d_out, int out_size, void* d_ws, size_t ws_size,
                              hipStream_t stream) {
  const float* in_hs     = (const float*)d_in[0];
  const float* in_c      = (const float*)d_in[1];
  const float* in_text   = (const float*)d_in[2];
  const float* in_normw  = (const float*)d_in[3];
  const float* in_adw    = (const float*)d_in[4];
  const float* in_adb    = (const float*)d_in[5];
  const float* in_inpw   = (const float*)d_in[6];
  const float* in_convw  = (const float*)d_in[7];
  const float* in_convb  = (const float*)d_in[8];
  const float* in_xpw    = (const float*)d_in[9];
  const float* in_dtw    = (const float*)d_in[10];
  const float* in_dtb    = (const float*)d_in[11];
  const float* in_alog   = (const float*)d_in[12];
  const float* in_ablog  = (const float*)d_in[13];
  const float* in_dfwd   = (const float*)d_in[14];
  const float* in_dbwd   = (const float*)d_in[15];
  const float* in_outw   = (const float*)d_in[16];
  const float* in_wq     = (const float*)d_in[17];
  const float* in_wk     = (const float*)d_in[18];
  const float* in_wv     = (const float*)d_in[19];
  const float* in_wo     = (const float*)d_in[20];
  const float* in_wob    = (const float*)d_in[21];

  // ---- workspace carve-up ----
  size_t off = 0;
  auto carve = [&](size_t bytes) -> char* {
    char* p = (char*)d_ws + off;
    off += (bytes + 255) & ~(size_t)255;
    return p;
  };
  float*          R    = (float*)carve((size_t)M_ * D_ * 4);
  float*          HS   = (float*)carve((size_t)M_ * D_ * 4);
  float*          HN   = (float*)carve((size_t)M_ * D_ * 4);
  unsigned short* HNB  = (unsigned short*)carve((size_t)M_ * D_ * 2);
  float*          XZ   = (float*)carve((size_t)M_ * 2 * DI_ * 4);
  float*          XC   = (float*)carve((size_t)M_ * DI_ * 4);
  unsigned short* XCB  = (unsigned short*)carve((size_t)M_ * DI_ * 2);
  float*          PROJ = (float*)carve((size_t)M_ * (DR_ + 2 * DS_) * 4);
  unsigned short* DTIN = (unsigned short*)carve((size_t)M_ * DR_ * 2);
  float*          DT   = (float*)carve((size_t)M_ * DI_ * 4);
  float*          Y    = (float*)carve((size_t)M_ * DI_ * 4);
  unsigned short* YB   = (unsigned short*)carve((size_t)M_ * DI_ * 2);
  float*          MIX  = (float*)carve((size_t)M_ * D_ * 4);
  float*          X    = (float*)carve((size_t)M_ * D_ * 4);
  float*          MOD  = (float*)carve((size_t)B_ * 6 * D_ * 4);
  unsigned short* TXB  = (unsigned short*)carve((size_t)M2_ * D_ * 2);
  float*          KB   = (float*)carve((size_t)M2_ * D_ * 4);
  float*          VB   = (float*)carve((size_t)M2_ * D_ * 4);
  float*          Q    = (float*)carve((size_t)B_ * D_ * 4);
  float*          OA   = (float*)carve((size_t)B_ * D_ * 4);
  float*          O2   = (float*)carve((size_t)B_ * D_ * 4);
  unsigned short* WIN  = (unsigned short*)carve((size_t)D_ * 2 * DI_ * 2);
  unsigned short* WXP  = (unsigned short*)carve((size_t)DI_ * (DR_ + 2 * DS_) * 2);
  unsigned short* WDT  = (unsigned short*)carve((size_t)DR_ * DI_ * 2);
  unsigned short* WOP  = (unsigned short*)carve((size_t)DI_ * D_ * 2);
  unsigned short* WKB  = (unsigned short*)carve((size_t)D_ * D_ * 2);
  unsigned short* WVB  = (unsigned short*)carve((size_t)D_ * D_ * 2);
  (void)ws_size; (void)n_in; (void)in_sizes; (void)out_size;

  auto cvt = [&](const float* s, unsigned short* d, int n) {
    cvt_bf16_kernel<<<ceil_div(n, 256), 256, 0, stream>>>(s, d, n);
  };

  const int mtiles  = ceil_div(M_, TM);   // 33
  const int m2tiles = ceil_div(M2_, TM);  // 5

  cvt(in_text, TXB, M2_ * D_);

  const float* hsPtr = in_hs;
  for (int layer = 0; layer < DEPTH_; layer++) {
    const float* normw = in_normw + layer * D_;
    const float* adw   = in_adw   + (size_t)layer * D_ * 6 * D_;
    const float* adb   = in_adb   + (size_t)layer * 6 * D_;
    const float* inpw  = in_inpw  + (size_t)layer * D_ * 2 * DI_;
    const float* convw = in_convw + (size_t)layer * DI_ * KK_;
    const float* convb = in_convb + (size_t)layer * DI_;
    const float* xpw   = in_xpw   + (size_t)layer * DI_ * (DR_ + 2 * DS_);
    const float* dtw   = in_dtw   + (size_t)layer * DR_ * DI_;
    const float* dtb   = in_dtb   + (size_t)layer * DI_;
    const float* alog  = in_alog  + (size_t)layer * DI_ * DS_;
    const float* ablog = in_ablog + (size_t)layer * DI_ * DS_;
    const float* dfwd  = in_dfwd  + (size_t)layer * DI_;
    const float* dbwd  = in_dbwd  + (size_t)layer * DI_;
    const float* outw  = in_outw  + (size_t)layer * DI_ * D_;
    const float* wq    = in_wq    + (size_t)layer * D_ * D_;
    const float* wk    = in_wk    + (size_t)layer * D_ * D_;
    const float* wv    = in_wv    + (size_t)layer * D_ * D_;
    const float* wo    = in_wo    + (size_t)layer * D_ * D_;
    const float* wob   = in_wob   + (size_t)layer * D_;

    // weights -> bf16 (must redo every call: no caching allowed)
    cvt(inpw, WIN, D_ * 2 * DI_);
    cvt(xpw,  WXP, DI_ * (DR_ + 2 * DS_));
    cvt(dtw,  WDT, DR_ * DI_);
    cvt(outw, WOP, DI_ * D_);
    cvt(wk,   WKB, D_ * D_);
    cvt(wv,   WVB, D_ * D_);

    prologue_kernel<<<M_, 256, 0, stream>>>(hsPtr, R, normw, HN, HNB, layer == 0);
    mod_kernel<<<B_ * 12, 256, 0, stream>>>(in_c, adw, adb, MOD);

    gemm_bf16_wmma<<<dim3(2 * DI_ / TN, mtiles), 256, 0, stream>>>(
        HNB, WIN, XZ, M_, 2 * DI_, D_);

    for (int dir = 0; dir < 2; dir++) {
      conv_silu_kernel<<<ceil_div(M_ * DI_, 256), 256, 0, stream>>>(
          XZ, convw, convb, XC, XCB, dir);
      gemm_bf16_wmma<<<dim3(1, mtiles), 256, 0, stream>>>(
          XCB, WXP, PROJ, M_, DR_ + 2 * DS_, DI_);
      prep_dtin_kernel<<<ceil_div(M_ * DR_, 256), 256, 0, stream>>>(PROJ, DTIN);
      gemm_bf16_wmma<<<dim3(DI_ / TN, mtiles), 256, 0, stream>>>(
          DTIN, WDT, DT, M_, DI_, DR_);
      softplus_kernel<<<ceil_div(M_ * DI_, 256), 256, 0, stream>>>(DT, dtb);
      scan_kernel<<<B_ * (DI_ / 256), 256, 0, stream>>>(
          XC, DT, PROJ, dir ? ablog : alog, dir ? dbwd : dfwd, Y, dir);
    }

    combine_kernel<<<ceil_div(M_ * DI_, 256), 256, 0, stream>>>(XZ, Y, YB);
    gemm_bf16_wmma<<<dim3(D_ / TN, mtiles), 256, 0, stream>>>(
        YB, WOP, MIX, M_, D_, DI_);
    xmod_kernel<<<ceil_div(M_ * D_, 256), 256, 0, stream>>>(HN, MIX, MOD, X);

    lnq_kernel<<<B_, 256, 0, stream>>>(X, MOD, wq, Q);
    gemm_bf16_wmma<<<dim3(D_ / TN, m2tiles), 256, 0, stream>>>(TXB, WKB, KB, M2_, D_, D_);
    gemm_bf16_wmma<<<dim3(D_ / TN, m2tiles), 256, 0, stream>>>(TXB, WVB, VB, M2_, D_, D_);
    attn_kernel<<<B_ * H_, 128, 0, stream>>>(Q, KB, VB, OA);
    oproj_kernel<<<B_ * 2, 256, 0, stream>>>(OA, wo, wob, O2);

    float* outPtr = (layer == DEPTH_ - 1) ? (float*)d_out : HS;
    epilogue_kernel<<<ceil_div(M_ * D_, 256), 256, 0, stream>>>(X, MOD, O2, outPtr);
    hsPtr = outPtr;
  }
}
